// RouterAttention_63711544869280
// MI455X (gfx1250) — matrix operand order repaired and checked by reading, not compiled, not run on hardware
//
#include <hip/hip_runtime.h>

// ---------------------------------------------------------------------------
// RouterAttention for MI455X (gfx1250, wave32, WMMA)
// Shapes: B=4, L=4096, D=512, H=8, E=64, R=64
// ---------------------------------------------------------------------------

typedef __bf16 bf16_t;
typedef __attribute__((ext_vector_type(16))) __bf16 v16bf;
typedef __attribute__((ext_vector_type(8)))  __bf16 v8bf;
typedef __attribute__((ext_vector_type(8)))  float  v8f;
typedef __attribute__((ext_vector_type(2)))  float  v2f;

static constexpr int BB   = 4;
static constexpr int LL   = 4096;
static constexpr int DD   = 512;
static constexpr int HH   = 8;
static constexpr int EE   = 64;
static constexpr int RR   = 64;
static constexpr int MR   = BB * LL;       // 16384 rows of x
static constexpr int NCAT = 4 * DD;        // 2048: q|k|skip|z
static constexpr int KCH  = 64;            // K-chunks for router_V split-K
#define INV_SQRT_E 0.125f
#define LOG_1E4    9.210340371976184f      // ln(10000)

// ---------------------------------------------------------------------------
// K0a: x (f32) -> xb (bf16)
// ---------------------------------------------------------------------------
__global__ __launch_bounds__(256) void k_cvt_x(const float* __restrict__ x,
                                               bf16_t* __restrict__ xb, int n) {
  int i = blockIdx.x * 256 + threadIdx.x;
  if (i < n) xb[i] = (bf16_t)x[i];
}

// ---------------------------------------------------------------------------
// K0b: pack W{q,k,skip,z} transposed -> wt[j][n][k] = W_j[k][n] (bf16)
// ---------------------------------------------------------------------------
__global__ __launch_bounds__(256) void k_pack_w(const float* __restrict__ Wq,
                                                const float* __restrict__ Wk,
                                                const float* __restrict__ Ws,
                                                const float* __restrict__ Wz,
                                                bf16_t* __restrict__ wt) {
  int id = blockIdx.x * 256 + threadIdx.x;           // 4*512*512 threads
  int j   = id >> 18;
  int idx = id & ((1 << 18) - 1);
  int n = idx >> 9;
  int k = idx & 511;
  const float* W = (j == 0) ? Wq : (j == 1) ? Wk : (j == 2) ? Ws : Wz;
  wt[(size_t)(j * DD + n) * DD + k] = (bf16_t)W[(size_t)k * DD + n];
}

// ---------------------------------------------------------------------------
// K0c: concat biases
// ---------------------------------------------------------------------------
__global__ __launch_bounds__(256) void k_pack_b(const float* __restrict__ bq,
                                                const float* __restrict__ bk,
                                                const float* __restrict__ bs,
                                                const float* __restrict__ bz,
                                                float* __restrict__ bc) {
  int id = blockIdx.x * 256 + threadIdx.x;           // 2048 threads
  int j = id >> 9, n = id & 511;
  const float* bsel = (j == 0) ? bq : (j == 1) ? bk : (j == 2) ? bs : bz;
  bc[id] = bsel[n];
}

// ---------------------------------------------------------------------------
// K1: fused projection GEMM  C[m, 0..2047] = x @ [Wq|Wk|Wskip|Wz] + bias
//     bf16 WMMA 16x16x32, f32 accumulate. SiLU on the z block (cols>=1536).
//     Block = 8 waves; wave computes a 16(M) x 64(N) strip; block = 128M x 64N.
//     Prefetches hoisted out of the k-loop so the scheduler can keep
//     s_wait_loadcnt > 0 (loads for tile t+1 in flight under WMMA t).
// ---------------------------------------------------------------------------
__global__ __launch_bounds__(256) void k_gemm_proj(const bf16_t* __restrict__ xb,
                                                   const bf16_t* __restrict__ wt,
                                                   const float* __restrict__ bc,
                                                   float* __restrict__ cbuf) {
  const int tid  = threadIdx.x;
  const int wave = tid >> 5;
  const int lane = tid & 31;
  const int l16  = lane & 15;
  const int kh   = lane >> 4;

  const int m0 = blockIdx.x * 128 + wave * 16;
  const int n0 = blockIdx.y * 64;

  v8f acc[4] = {v8f{0}, v8f{0}, v8f{0}, v8f{0}};

  const bf16_t* arow = xb + (size_t)(m0 + l16) * DD;
  // One-time warm of this wave's A row (1 KB): stays out of the inner loop.
  __builtin_prefetch(arow, 0, 1);                    // global_prefetch_b8
  __builtin_prefetch(arow + 256, 0, 1);

#pragma unroll 4
  for (int kk = 0; kk < DD; kk += 32) {
    union { v16bf v; v8bf h[2]; } A;
    A.h[0] = *(const v8bf*)(arow + kk + 8 * kh);        // K = kk +  8h + 0..7
    A.h[1] = *(const v8bf*)(arow + kk + 16 + 8 * kh);   // K = kk + 16 + 8h + 0..7
#pragma unroll
    for (int t = 0; t < 4; ++t) {
      const bf16_t* brow = wt + (size_t)(n0 + 16 * t + l16) * DD;
      union { v16bf v; v8bf h[2]; } Bf;                 // K = kk + 8h + 0..7, kk + 16 + 8h + 0..7
      Bf.h[0] = *(const v8bf*)(brow + kk + 8 * kh);
      Bf.h[1] = *(const v8bf*)(brow + kk + 16 + 8 * kh);
      acc[t] = __builtin_amdgcn_wmma_f32_16x16x32_bf16(
          false, A.v, false, Bf.v, (short)0, acc[t], false, false);
    }
  }

#pragma unroll
  for (int t = 0; t < 4; ++t) {
    const int n = n0 + 16 * t + l16;
    const float bias = bc[n];
#pragma unroll
    for (int j = 0; j < 8; ++j) {
      const int m = m0 + j + 8 * kh;
      float v = acc[t][j] + bias;
      if (n >= 3 * DD) v = v * (1.0f / (1.0f + __expf(-v)));   // SiLU for z
      cbuf[(size_t)m * NCAT + n] = v;
    }
  }
}

// ---------------------------------------------------------------------------
// K2: router scores  S[bh][r][s] = scale * dot(r_h[r], k_h[s])  (f32 WMMA)
//     grid: (L/32, B*H); block 8 waves -> 4 r-tiles x 2 s-tiles
// ---------------------------------------------------------------------------
__global__ __launch_bounds__(256) void k_router_scores(const float* __restrict__ rproj,
                                                       const float* __restrict__ cbuf,
                                                       float* __restrict__ sbuf) {
  const int tid  = threadIdx.x;
  const int wave = tid >> 5;
  const int lane = tid & 31;
  const int l16  = lane & 15;
  const int kh   = lane >> 4;
  const int rt   = wave & 3;
  const int st   = wave >> 2;

  const int bh = blockIdx.y;
  const int b  = bh >> 3, h = bh & 7;
  const int s  = blockIdx.x * 32 + 16 * st + l16;

  const float* rrow = rproj + (size_t)(16 * rt + l16) * DD + h * EE;
  const float* krow = cbuf + ((size_t)(b * LL + s)) * NCAT + DD + h * EE;

  v8f acc = v8f{0};
#pragma unroll
  for (int kk = 0; kk < EE; kk += 4) {
    v2f a; a[0] = rrow[kk + 2 * kh]; a[1] = rrow[kk + 2 * kh + 1];
    v2f bb; bb[0] = krow[kk + 2 * kh]; bb[1] = krow[kk + 2 * kh + 1];
    acc = __builtin_amdgcn_wmma_f32_16x16x4_f32(
        false, a, false, bb, (short)0, acc, false, false);
  }
#pragma unroll
  for (int j = 0; j < 8; ++j) {
    const int rr = 16 * rt + j + 8 * kh;
    sbuf[((size_t)bh * RR + rr) * LL + s] = acc[j] * INV_SQRT_E;
  }
}

// ---------------------------------------------------------------------------
// K3: per-row softmax stats over s (4096): max and sum(exp)
// ---------------------------------------------------------------------------
__global__ __launch_bounds__(256) void k_row_stats(const float* __restrict__ sbuf,
                                                   float* __restrict__ rmax,
                                                   float* __restrict__ rsum) {
  __shared__ float red[256];
  const int tid = threadIdx.x;
  const int row = blockIdx.x;
  const float* p = sbuf + (size_t)row * LL;

  float m = -3.4e38f;
  for (int i = tid; i < LL; i += 256) m = fmaxf(m, p[i]);
  red[tid] = m; __syncthreads();
  for (int s = 128; s > 0; s >>= 1) {
    if (tid < s) red[tid] = fmaxf(red[tid], red[tid + s]);
    __syncthreads();
  }
  const float mx = red[0]; __syncthreads();

  float sum = 0.0f;
  for (int i = tid; i < LL; i += 256) sum += __expf(p[i] - mx);
  red[tid] = sum; __syncthreads();
  for (int s = 128; s > 0; s >>= 1) {
    if (tid < s) red[tid] += red[tid + s];
    __syncthreads();
  }
  if (tid == 0) { rmax[row] = mx; rsum[row] = red[0]; }
}

// ---------------------------------------------------------------------------
// K4: normalize + reverse-RoPE rotate row pairs (2k,2k+1) in place
// ---------------------------------------------------------------------------
__global__ __launch_bounds__(256) void k_rope_rows(float* __restrict__ sbuf,
                                                   const float* __restrict__ rmax,
                                                   const float* __restrict__ rsum) {
  const int id = blockIdx.x * 256 + threadIdx.x;    // 32 * 32 * 4096 threads
  const int s  = id & (LL - 1);
  const int r2 = id >> 12;
  const int kp = r2 & 31;
  const int bh = r2 >> 5;
  const int row0 = bh * RR + 2 * kp;

  float* p0 = sbuf + (size_t)row0 * LL + s;
  float* p1 = p0 + LL;
  const float e0 = __expf(*p0 - rmax[row0]) / rsum[row0];
  const float e1 = __expf(*p1 - rmax[row0 + 1]) / rsum[row0 + 1];

  const float pos   = (float)(LL - 1 - s);
  const float theta = __expf(-(float)kp * (LOG_1E4 / 32.0f));
  float sn, cs; __sincosf(pos * theta, &sn, &cs);
  *p0 = e0 * cs - e1 * sn;
  *p1 = e0 * sn + e1 * cs;
}

// ---------------------------------------------------------------------------
// K5a: split-K partial router_V: part[bh][chunk] = A''[:,chunk] @ v[chunk,:]
//      grid (64 chunks, 32 bh); A'' and v tiles staged in LDS (rows padded
//      to 66 floats to spread the strided A-fragment reads across banks).
//      8 waves x 2 output tiles x 16 k-steps = 32 WMMAs per wave from LDS.
// ---------------------------------------------------------------------------
__global__ __launch_bounds__(256) void k_router_v_part(const float* __restrict__ sbuf,
                                                       const float* __restrict__ x,
                                                       float* __restrict__ part) {
  __shared__ float As[64 * 66];
  __shared__ float Vs[64 * 66];
  const int tid  = threadIdx.x;
  const int wave = tid >> 5;
  const int lane = tid & 31;
  const int l16  = lane & 15;
  const int kh   = lane >> 4;

  const int chunk = blockIdx.x;                  // 0..63, s-range [64*chunk, +64)
  const int bh    = blockIdx.y;
  const int b     = bh >> 3, h = bh & 7;
  const int s0    = chunk * 64;

  for (int i = tid; i < 64 * 64; i += 256) {     // A'' tile: 64 r x 64 s
    const int r = i >> 6, c = i & 63;
    As[r * 66 + c] = sbuf[((size_t)bh * RR + r) * LL + s0 + c];
  }
  for (int i = tid; i < 64 * 64; i += 256) {     // v tile: 64 s x 64 d
    const int s = i >> 6, d = i & 63;
    Vs[s * 66 + d] = x[((size_t)(b * LL + s0 + s)) * DD + h * EE + d];
  }
  __syncthreads();

  for (int i = 0; i < 2; ++i) {
    const int t  = wave * 2 + i;
    const int mt = t >> 2, nt = t & 3;
    v8f acc = v8f{0};
#pragma unroll
    for (int kk = 0; kk < 64; kk += 4) {
      const int k0 = kk + 2 * kh;
      v2f a;  a[0]  = As[(16 * mt + l16) * 66 + k0];
              a[1]  = As[(16 * mt + l16) * 66 + k0 + 1];
      v2f bb; bb[0] = Vs[k0 * 66 + 16 * nt + l16];
              bb[1] = Vs[(k0 + 1) * 66 + 16 * nt + l16];
      acc = __builtin_amdgcn_wmma_f32_16x16x4_f32(
          false, a, false, bb, (short)0, acc, false, false);
    }
#pragma unroll
    for (int j = 0; j < 8; ++j) {
      const int m = 16 * mt + j + 8 * kh;
      const int n = 16 * nt + l16;
      part[((size_t)bh * KCH + chunk) * (RR * EE) + m * EE + n] = acc[j];
    }
  }
}

// ---------------------------------------------------------------------------
// K5b: deterministic reduce of split-K partials -> rv[bh][r][d]
// ---------------------------------------------------------------------------
__global__ __launch_bounds__(256) void k_router_v_reduce(const float* __restrict__ part,
                                                         float* __restrict__ rv) {
  const int id = blockIdx.x * 256 + threadIdx.x;   // 32 * 4096
  const int bh = id >> 12;
  const int e  = id & (RR * EE - 1);
  const float* p = part + (size_t)bh * KCH * (RR * EE) + e;
  float s = 0.0f;
#pragma unroll 8
  for (int c = 0; c < KCH; ++c) s += p[(size_t)c * (RR * EE)];
  rv[(size_t)bh * (RR * EE) + e] = s;
}

// ---------------------------------------------------------------------------
// K6: q_scores[bh][l][r] = scale * dot(q_h[l], r_h[r])  (f32 WMMA)
//     Writes into the (now consumed) scores buffer, layout (bh, l, 64).
// ---------------------------------------------------------------------------
__global__ __launch_bounds__(256) void k_q_scores(const float* __restrict__ cbuf,
                                                  const float* __restrict__ rproj,
                                                  float* __restrict__ qs) {
  const int tid  = threadIdx.x;
  const int wave = tid >> 5;
  const int lane = tid & 31;
  const int l16  = lane & 15;
  const int kh   = lane >> 4;
  const int lt   = wave >> 2;
  const int rt   = wave & 3;

  const int bh = blockIdx.y;
  const int b  = bh >> 3, h = bh & 7;
  const int l0 = blockIdx.x * 32 + 16 * lt;

  const float* qrow = cbuf + ((size_t)(b * LL + l0 + l16)) * NCAT + h * EE;
  const float* rrow = rproj + (size_t)(16 * rt + l16) * DD + h * EE;

  v8f acc = v8f{0};
#pragma unroll
  for (int kk = 0; kk < EE; kk += 4) {
    v2f a; a[0] = qrow[kk + 2 * kh]; a[1] = qrow[kk + 2 * kh + 1];
    v2f bb; bb[0] = rrow[kk + 2 * kh]; bb[1] = rrow[kk + 2 * kh + 1];
    acc = __builtin_amdgcn_wmma_f32_16x16x4_f32(
        false, a, false, bb, (short)0, acc, false, false);
  }
#pragma unroll
  for (int j = 0; j < 8; ++j) {
    const int lrow = l0 + j + 8 * kh;
    const int rcol = 16 * rt + l16;
    qs[((size_t)bh * LL + lrow) * RR + rcol] = acc[j] * INV_SQRT_E;
  }
}

// ---------------------------------------------------------------------------
// K7: per-row tail: softmax(64) + reverse-RoPE + (a @ RV) + skip, * z
//     One wave per (b,h,l) row; 8 rows per block; RV cached in LDS.
// ---------------------------------------------------------------------------
__global__ __launch_bounds__(256) void k_final(const float* __restrict__ qs,
                                               const float* __restrict__ rv,
                                               const float* __restrict__ cbuf,
                                               float* __restrict__ out) {
  __shared__ float rvs[RR * EE];                  // 16 KB
  const int tid  = threadIdx.x;
  const int wave = tid >> 5;
  const int lane = tid & 31;

  const int row0 = blockIdx.x * 8;                // 8 rows, all same bh (4096%8==0)
  const int bh   = row0 >> 12;
  const int b    = bh >> 3, h = bh & 7;

  const float* rvsrc = rv + (size_t)bh * RR * EE;
  for (int i = tid; i < RR * EE; i += 256) rvs[i] = rvsrc[i];
  __syncthreads();

  const int row = row0 + wave;
  const int l   = row & (LL - 1);

  const float* arow = qs + (size_t)row * RR;
  float a0 = arow[2 * lane], a1 = arow[2 * lane + 1];

  float mx = fmaxf(a0, a1);
  for (int off = 16; off > 0; off >>= 1) mx = fmaxf(mx, __shfl_xor(mx, off, 32));
  float e0 = __expf(a0 - mx), e1 = __expf(a1 - mx);
  float z = e0 + e1;
  for (int off = 16; off > 0; off >>= 1) z += __shfl_xor(z, off, 32);
  e0 /= z; e1 /= z;

  const float pos   = (float)(LL - 1 - l);
  const float theta = __expf(-(float)lane * (LOG_1E4 / 32.0f));
  float sn, cs; __sincosf(pos * theta, &sn, &cs);
  const float p0 = e0 * cs - e1 * sn;
  const float p1 = e0 * sn + e1 * cs;

  float acc0 = 0.0f, acc1 = 0.0f;
#pragma unroll 8
  for (int r = 0; r < RR; ++r) {
    const float av = __shfl((r & 1) ? p1 : p0, r >> 1, 32);
    acc0 += av * rvs[r * EE + lane];
    acc1 += av * rvs[r * EE + lane + 32];
  }

  const size_t base = ((size_t)(b * LL + l)) * NCAT + h * EE;
  const float sk0 = cbuf[base + 2 * DD + lane];
  const float sk1 = cbuf[base + 2 * DD + lane + 32];
  const float zg0 = cbuf[base + 3 * DD + lane];
  const float zg1 = cbuf[base + 3 * DD + lane + 32];

  const size_t ob = ((size_t)(b * LL + l)) * DD + h * EE;
  out[ob + lane]      = (acc0 + sk0) * zg0;
  out[ob + lane + 32] = (acc1 + sk1) * zg1;
}

// ---------------------------------------------------------------------------
// Host-side launcher
// ---------------------------------------------------------------------------
extern "C" void kernel_launch(void* const* d_in, const int* in_sizes, int n_in,
                              void* d_out, int out_size, void* d_ws, size_t ws_size,
                              hipStream_t stream) {
  (void)in_sizes; (void)n_in; (void)out_size; (void)ws_size;

  const float* x     = (const float*)d_in[0];
  const float* Wq    = (const float*)d_in[1];
  const float* bq    = (const float*)d_in[2];
  const float* Wk    = (const float*)d_in[3];
  const float* bk    = (const float*)d_in[4];
  const float* Wskip = (const float*)d_in[5];
  const float* bskip = (const float*)d_in[6];
  const float* Wz    = (const float*)d_in[7];
  const float* bz    = (const float*)d_in[8];
  const float* rproj = (const float*)d_in[9];
  float* out = (float*)d_out;

  char* ws = (char*)d_ws;
  const size_t off_xb = 0;
  const size_t off_wt = off_xb + (size_t)MR * DD * sizeof(bf16_t);      // 16 MB
  const size_t off_bc = off_wt + (size_t)4 * DD * DD * sizeof(bf16_t);  // +2 MB
  const size_t off_c  = (off_bc + (size_t)NCAT * sizeof(float) + 255) & ~(size_t)255;
  const size_t off_s  = off_c + (size_t)MR * NCAT * sizeof(float);      // +128 MB
  const size_t off_mx = off_s + (size_t)32 * RR * LL * sizeof(float);   // +32 MB
  const size_t off_sm = off_mx + (size_t)32 * RR * sizeof(float);
  const size_t off_rv = (off_sm + (size_t)32 * RR * sizeof(float) + 255) & ~(size_t)255;
  const size_t off_pp = off_rv + (size_t)32 * RR * EE * sizeof(float);  // partials
                                                                        // +33.5 MB

  bf16_t* xb   = (bf16_t*)(ws + off_xb);
  bf16_t* wt   = (bf16_t*)(ws + off_wt);
  float*  bc   = (float*)(ws + off_bc);
  float*  cbuf = (float*)(ws + off_c);
  float*  sbuf = (float*)(ws + off_s);
  float*  rmax = (float*)(ws + off_mx);
  float*  rsum = (float*)(ws + off_sm);
  float*  rv   = (float*)(ws + off_rv);
  float*  part = (float*)(ws + off_pp);

  // K0: packing
  k_cvt_x<<<(MR * DD) / 256, 256, 0, stream>>>(x, xb, MR * DD);
  k_pack_w<<<(4 * DD * DD) / 256, 256, 0, stream>>>(Wq, Wk, Wskip, Wz, wt);
  k_pack_b<<<NCAT / 256, 256, 0, stream>>>(bq, bk, bskip, bz, bc);

  // K1: fused q|k|skip|silu(z) projection GEMM (bf16 WMMA)
  k_gemm_proj<<<dim3(MR / 128, NCAT / 64), 256, 0, stream>>>(xb, wt, bc, cbuf);

  // K2-K5: router side
  k_router_scores<<<dim3(LL / 32, BB * HH), 256, 0, stream>>>(rproj, cbuf, sbuf);
  k_row_stats<<<32 * RR, 256, 0, stream>>>(sbuf, rmax, rsum);
  k_rope_rows<<<(32 * 32 * LL) / 256, 256, 0, stream>>>(sbuf, rmax, rsum);
  k_router_v_part<<<dim3(KCH, BB * HH), 256, 0, stream>>>(sbuf, x, part);
  k_router_v_reduce<<<(32 * RR * EE) / 256, 256, 0, stream>>>(part, rv);

  // K6: q scores (reuses sbuf region, stream-serialized after K5)
  k_q_scores<<<dim3(LL / 32, BB * HH), 256, 0, stream>>>(cbuf, rproj, sbuf);

  // K7: softmax + RoPE + GEMV + skip/z epilogue
  k_final<<<(32 * LL) / 8, 256, 0, stream>>>(sbuf, rv, cbuf, out);
}
